// LightDecoder_35579509080403
// MI455X (gfx1250) — matrix operand order repaired and checked by reading, not compiled, not run on hardware
//
#include <hip/hip_runtime.h>
#include <hip/hip_bf16.h>

// ---------------------------------------------------------------------------
// DSVT encoder stack for MI455X (gfx1250, wave32, WMMA).
//  * gather/scatter are row permutations commuting with row GEMMs ->
//    projections/FFN are dense bf16 WMMA GEMMs reading fragments directly
//    from global (A = two 16B row chunks, B = contiguous 32B W-row chunk).
//  * per-set attention core (36x36, 8 heads) also runs on WMMA: pad M 36->48,
//    K 36->64; V transposed into bank-conflict-free padded LDS; mask folded
//    into -inf score bias so padding vanishes through softmax.
//  * Q/K set gather uses CDNA5 async global->LDS loads (ASYNCcnt).
// ---------------------------------------------------------------------------

typedef __bf16 bf;
typedef __attribute__((ext_vector_type(8)))  __bf16 v8bf;
typedef __attribute__((ext_vector_type(16))) __bf16 v16bf;
typedef __attribute__((ext_vector_type(8)))  float  v8f;

#define NV    262152   // voxels
#define CCH   256      // d_model
#define DFF_  1024
#define NSET_ 7282
#define SETSZ 36
#define LAYERS 4

__device__ __forceinline__ void async_ld_b128(void* lds_dst, const void* gsrc)
{
    unsigned l = (unsigned)(size_t)lds_dst;           // low 32 bits = LDS offset
    asm volatile("global_load_async_to_lds_b128 %0, %1, off"
                 :: "v"(l), "v"(gsrc) : "memory");
}
__device__ __forceinline__ void wait_async0()
{
    asm volatile("s_wait_asynccnt 0x0" ::: "memory");
}

// ---------------------------------------------------------------------------
// bf16 WMMA GEMM:  Out[M x O] = A[M x K] @ W[O x K]^T (+bias) (opt GELU)
// 256 thr = 8 waves; wave w -> rows [blk*128 + w*16, +16), 64 cols.
// ---------------------------------------------------------------------------
template<bool GELU, bool F32OUT, bool BF16OUT>
__global__ __launch_bounds__(256)
void gemm_bf16_wmma(const bf* __restrict__ A, int lda,
                    const bf* __restrict__ W, int ldw,
                    const float* __restrict__ bias,
                    float* __restrict__ Of, bf* __restrict__ Ob, int ldo,
                    int M, int K)
{
    const int wave = threadIdx.x >> 5;
    const int lane = threadIdx.x & 31;
    const int half = lane >> 4;
    const int l16  = lane & 15;
    const int rowBase = blockIdx.x * 128 + wave * 16;
    const int colBase = blockIdx.y * 64;

    int arow = rowBase + l16;
    if (arow >= M) arow = M - 1;                 // clamp: stores are guarded
    const bf* ap  = A + (size_t)arow * lda + half * 16;
    const bf* wp0 = W + (size_t)(colBase + l16) * ldw + half * 16;

    v8f acc[4];
#pragma unroll
    for (int t = 0; t < 4; ++t) acc[t] = (v8f)(0.0f);

    for (int k0 = 0; k0 < K; k0 += 32) {
        v8bf a0 = *(const v8bf*)(ap + k0);        // K  0.. 7 (half0) /16..23
        v8bf a1 = *(const v8bf*)(ap + k0 + 8);    // K  8..15 (half0) /24..31
        v16bf afrag;
#pragma unroll
        for (int i = 0; i < 8; ++i) { afrag[i] = a0[i]; afrag[8 + i] = a1[i]; }
#pragma unroll
        for (int t = 0; t < 4; ++t) {
            v16bf bfrag = *(const v16bf*)(wp0 + (size_t)(t * 16) * ldw + k0);
            acc[t] = __builtin_amdgcn_wmma_f32_16x16x32_bf16(
                         false, afrag, false, bfrag,
                         (short)0, acc[t], false, false);
        }
    }

#pragma unroll
    for (int t = 0; t < 4; ++t) {
        const int col = colBase + t * 16 + l16;
        const float bv = bias ? bias[col] : 0.0f;
#pragma unroll
        for (int g = 0; g < 8; ++g) {
            const int row = rowBase + g + half * 8;   // C/D VGPR layout
            if (row < M) {
                float v = acc[t][g] + bv;
                if (GELU) v = 0.5f * v * (1.0f + erff(v * 0.70710678118654752f));
                if (F32OUT)  Of[(size_t)row * ldo + col] = v;
                if (BF16OUT) Ob[(size_t)row * ldo + col] = (bf)v;
            }
        }
    }
}

// ---------------------------------------------------------------------------
// Per-set masked MHA core, WMMA edition. 1 block/set, 8 waves = 8 heads.
//   phase 1: async-gather Q,K rows (stride-264 LDS, bank step 4), build
//            zero-padded V^T (stride 72 -> bank step 36, 16B aligned rows)
//   phase 2: scores S = Q K^T : 9 x wmma(16x16x32) per head, +mask/-inf pad
//   phase 3: softmax over 48 cols -> bf16 probs (stride 72, K-pad zeroed),
//            overlaying the dead Q/K region
//   phase 4: O = P V : 12 x wmma per head, scatter rows to attn buffer
// ---------------------------------------------------------------------------
#define OFF_V  41472                       // region A (qs+ks | pb) = 41472 B
#define OFF_S  (41472 + 36864)             // vsT = 256*72*2
#define OFF_R  (OFF_S + 56448)             // sc = 8*36*49*4
#define OFF_M  (OFF_R + 160)
#define SMEMSZ (OFF_M + 192)

__global__ __launch_bounds__(256)
void attn_kernel(const bf* __restrict__ qkbuf, const bf* __restrict__ vbuf,
                 const int* __restrict__ inds, const unsigned char* __restrict__ masks,
                 bf* __restrict__ attn)
{
    __shared__ __align__(16) char smem[SMEMSZ];
    bf*    qs   = (bf*)smem;               // 36 x 264
    bf*    ks   = (bf*)smem + 36 * 264;    // 36 x 264
    bf*    pb   = (bf*)smem;               // 8 x 36 x 72 (overlays qs/ks)
    bf*    vsT  = (bf*)(smem + OFF_V);     // 256 x 72 (keys padded to 64, zeroed)
    float* sc   = (float*)(smem + OFF_S);  // 8 x 36 x 49
    int*   ridx = (int*)(smem + OFF_R);
    float* mneg = (float*)(smem + OFF_M);  // 48 (-inf for key pad 36..47)

    const int s = blockIdx.x, tid = threadIdx.x;

    if (tid < SETSZ) ridx[tid] = inds[s * SETSZ + tid];
    if (tid < 48) {
        float m = -__builtin_inff();
        if (tid < SETSZ) m = masks[s * SETSZ + tid] ? -__builtin_inff() : 0.0f;
        mneg[tid] = m;
    }
    // zero V^T (pad keys must be exactly 0)
    for (int u = tid; u < 256 * 72 / 8; u += 256)
        ((uint4*)vsT)[u] = make_uint4(0u, 0u, 0u, 0u);
    __syncthreads();

    // async gather Q,K rows of this set straight into LDS
    for (int u = tid; u < SETSZ * 32; u += 256) {
        const int r = u >> 5, seg = u & 31;
        const bf* gq = qkbuf + (size_t)ridx[r] * 512 + seg * 8;
        async_ld_b128(qs + r * 264 + seg * 8, gq);
        async_ld_b128(ks + r * 264 + seg * 8, gq + 256);
    }
    // transposed V fill: thread = channel, loop keys (coalesced global reads)
    for (int k = 0; k < SETSZ; ++k)
        vsT[tid * 72 + k] = vbuf[(size_t)ridx[k] * 256 + tid];
    wait_async0();
    __syncthreads();

    const int h = tid >> 5, lane = tid & 31, half = lane >> 4, l16 = lane & 15;
    const int off = h * 32;
    float* sch = sc + h * 36 * 49;
    const float scale = 0.17677669529663687f;   // 1/sqrt(32)

    // ---- scores ----
#pragma unroll
    for (int mt = 0; mt < 3; ++mt) {
        int qrow = mt * 16 + l16; if (qrow > 35) qrow = 35;   // pad rows dup
        v8bf a0 = *(const v8bf*)(qs + qrow * 264 + off + half * 16);
        v8bf a1 = *(const v8bf*)(qs + qrow * 264 + off + 8 + half * 16);
        v16bf af;
#pragma unroll
        for (int i = 0; i < 8; ++i) { af[i] = a0[i]; af[8 + i] = a1[i]; }
#pragma unroll
        for (int nt = 0; nt < 3; ++nt) {
            int krow = nt * 16 + l16; if (krow > 35) krow = 35;
            v16bf bfr = *(const v16bf*)(ks + krow * 264 + off + half * 16);
            v8f d = (v8f)(0.0f);
            d = __builtin_amdgcn_wmma_f32_16x16x32_bf16(
                    false, af, false, bfr, (short)0, d, false, false);
            const int kj = nt * 16 + l16;
            const float mb = mneg[kj];
#pragma unroll
            for (int g = 0; g < 8; ++g) {
                const int qi = mt * 16 + g + half * 8;
                if (qi < SETSZ) sch[qi * 49 + kj] = d[g] * scale + mb;
            }
        }
    }
    __syncthreads();

    // ---- softmax over 48 key slots (pad slots are -inf -> prob 0) ----
    bf* pbh = pb + h * 36 * 72;
    for (int qi = lane; qi < SETSZ; qi += 32) {
        float* row = sch + qi * 49;
        float mx = -__builtin_inff();
        for (int j = 0; j < 48; ++j) mx = fmaxf(mx, row[j]);
        float sum = 0.0f;
        for (int j = 0; j < 48; ++j) { float e = __expf(row[j] - mx); row[j] = e; sum += e; }
        const float inv = 1.0f / sum;
        bf* prow = pbh + qi * 72;
        for (int j = 0; j < 48; ++j) prow[j] = (bf)(row[j] * inv);
        for (int j = 48; j < 64; ++j) prow[j] = (bf)0.0f;     // K-pad
    }
    __syncthreads();

    // ---- O = P V ----
#pragma unroll
    for (int mt = 0; mt < 3; ++mt) {
        int prow = mt * 16 + l16; if (prow > 35) prow = 35;
        v16bf af[2];
#pragma unroll
        for (int kc = 0; kc < 2; ++kc) {
            v8bf a0 = *(const v8bf*)(pbh + prow * 72 + kc * 32 + half * 16);
            v8bf a1 = *(const v8bf*)(pbh + prow * 72 + kc * 32 + 8 + half * 16);
#pragma unroll
            for (int i = 0; i < 8; ++i) { af[kc][i] = a0[i]; af[kc][8 + i] = a1[i]; }
        }
#pragma unroll
        for (int nt = 0; nt < 2; ++nt) {
            const int ch = off + nt * 16 + l16;
            v8f d = (v8f)(0.0f);
#pragma unroll
            for (int kc = 0; kc < 2; ++kc) {
                v16bf bfr = *(const v16bf*)(vsT + ch * 72 + kc * 32 + half * 16);
                d = __builtin_amdgcn_wmma_f32_16x16x32_bf16(
                        false, af[kc], false, bfr, (short)0, d, false, false);
            }
#pragma unroll
            for (int g = 0; g < 8; ++g) {
                const int qi = mt * 16 + g + half * 8;
                if (qi < SETSZ)
                    attn[(size_t)ridx[qi] * 256 + ch] = (bf)d[g];
            }
        }
    }
}

// ---------------------------------------------------------------------------
// y = LayerNorm(a + b) * g + beta ; one wave per 256-wide row, 8 rows/block.
// ---------------------------------------------------------------------------
__global__ __launch_bounds__(256)
void ln_add_kernel(const float* __restrict__ a, const float* __restrict__ b,
                   const float* __restrict__ g, const float* __restrict__ beta,
                   float* __restrict__ of, bf* __restrict__ ob, int M)
{
    const int row = blockIdx.x * 8 + (threadIdx.x >> 5);
    if (row >= M) return;
    const int lane = threadIdx.x & 31;
    const float4* pa = (const float4*)(a + (size_t)row * CCH);
    const float4* pb = (const float4*)(b + (size_t)row * CCH);
    float4 a0 = pa[lane * 2], a1 = pa[lane * 2 + 1];
    float4 b0 = pb[lane * 2], b1 = pb[lane * 2 + 1];
    float v[8] = {a0.x + b0.x, a0.y + b0.y, a0.z + b0.z, a0.w + b0.w,
                  a1.x + b1.x, a1.y + b1.y, a1.z + b1.z, a1.w + b1.w};
    float s = 0.0f, ss = 0.0f;
#pragma unroll
    for (int i = 0; i < 8; ++i) { s += v[i]; ss += v[i] * v[i]; }
#pragma unroll
    for (int m = 16; m >= 1; m >>= 1) {
        s  += __shfl_xor(s,  m, 32);
        ss += __shfl_xor(ss, m, 32);
    }
    const float mean = s * (1.0f / 256.0f);
    const float var  = ss * (1.0f / 256.0f) - mean * mean;
    const float rstd = rsqrtf(var + 1e-5f);
    const int c0 = lane * 8;
    float o[8];
#pragma unroll
    for (int i = 0; i < 8; ++i) o[i] = (v[i] - mean) * rstd * g[c0 + i] + beta[c0 + i];
    float4* po = (float4*)(of + (size_t)row * CCH);
    po[lane * 2]     = make_float4(o[0], o[1], o[2], o[3]);
    po[lane * 2 + 1] = make_float4(o[4], o[5], o[6], o[7]);
    if (ob) {
        bf* pob = ob + (size_t)row * CCH + c0;
#pragma unroll
        for (int i = 0; i < 8; ++i) pob[i] = (bf)o[i];
    }
}

__global__ __launch_bounds__(256)
void prep_kernel(const float* __restrict__ x, const float* __restrict__ pos,
                 bf* __restrict__ xq, bf* __restrict__ xb, int n)
{
    const int i = blockIdx.x * 256 + threadIdx.x;
    if (i < n) {
        const float xv = x[i];
        xq[i] = (bf)(xv + pos[i]);
        xb[i] = (bf)xv;
    }
}

__global__ __launch_bounds__(256)
void cvt_f32_bf16_kernel(const float* __restrict__ in, bf* __restrict__ out, int n)
{
    const int i = blockIdx.x * 256 + threadIdx.x;
    if (i < n) out[i] = (bf)in[i];
}

// ---------------------------------------------------------------------------
extern "C" void kernel_launch(void* const* d_in, const int* in_sizes, int n_in,
                              void* d_out, int out_size, void* d_ws, size_t ws_size,
                              hipStream_t stream)
{
    (void)in_sizes; (void)n_in; (void)out_size; (void)ws_size;
    const float* src       = (const float*)d_in[0];
    const float* pos       = (const float*)d_in[1];
    const int*   inds      = (const int*)d_in[2];
    const unsigned char* masks = (const unsigned char*)d_in[3];
    const float* in_proj_w = (const float*)d_in[4];
    const float* in_proj_b = (const float*)d_in[5];
    const float* out_w     = (const float*)d_in[6];
    const float* out_b     = (const float*)d_in[7];
    const float* lin1_w    = (const float*)d_in[8];
    const float* lin1_b    = (const float*)d_in[9];
    const float* lin2_w    = (const float*)d_in[10];
    const float* lin2_b    = (const float*)d_in[11];
    const float* n1_g      = (const float*)d_in[12];
    const float* n1_b      = (const float*)d_in[13];
    const float* n2_g      = (const float*)d_in[14];
    const float* n2_b      = (const float*)d_in[15];
    const float* enc_g     = (const float*)d_in[16];
    const float* enc_b     = (const float*)d_in[17];
    const float* res_g     = (const float*)d_in[18];
    const float* res_b     = (const float*)d_in[19];

    // ---- carve scratch ----
    char* ws = (char*)d_ws;
    size_t off = 0;
    auto carve = [&](size_t bytes) -> void* {
        void* p = ws + off; off += (bytes + 255) & ~(size_t)255; return p;
    };
    bf*    wqkv    = (bf*)carve((size_t)LAYERS * 768 * CCH * 2);
    bf*    wo      = (bf*)carve((size_t)LAYERS * CCH * CCH * 2);
    bf*    w1      = (bf*)carve((size_t)LAYERS * DFF_ * CCH * 2);
    bf*    w2      = (bf*)carve((size_t)LAYERS * CCH * DFF_ * 2);
    float* x       = (float*)carve((size_t)NV * CCH * 4);
    float* ident   = (float*)carve((size_t)NV * CCH * 4);
    float* resid   = (float*)carve((size_t)NV * CCH * 4);
    float* projf   = (float*)carve((size_t)NV * CCH * 4);
    bf*    xq      = (bf*)carve((size_t)NV * CCH * 2);
    bf*    xb      = (bf*)carve((size_t)NV * CCH * 2);
    bf*    qkb     = (bf*)carve((size_t)NV * 512 * 2);
    bf*    vb      = (bf*)carve((size_t)NV * CCH * 2);
    bf*    attnb   = (bf*)carve((size_t)NV * CCH * 2);
    bf*    hb      = (bf*)carve((size_t)NV * DFF_ * 2);

    const size_t NB  = (size_t)NV * CCH * sizeof(float);
    const int    nel = NV * CCH;
    const int    EG  = (nel + 255) / 256;   // elementwise grid
    const int    MB  = (NV + 127) / 128;    // gemm row blocks
    const int    LG  = (NV + 7) / 8;        // layernorm grid

    // ---- weights -> bf16 (once per launch, deterministic) ----
    cvt_f32_bf16_kernel<<<(LAYERS * 768 * CCH + 255) / 256, 256, 0, stream>>>(in_proj_w, wqkv, LAYERS * 768 * CCH);
    cvt_f32_bf16_kernel<<<(LAYERS * CCH * CCH + 255) / 256, 256, 0, stream>>>(out_w, wo, LAYERS * CCH * CCH);
    cvt_f32_bf16_kernel<<<(LAYERS * DFF_ * CCH + 255) / 256, 256, 0, stream>>>(lin1_w, w1, LAYERS * DFF_ * CCH);
    cvt_f32_bf16_kernel<<<(LAYERS * CCH * DFF_ + 255) / 256, 256, 0, stream>>>(lin2_w, w2, LAYERS * CCH * DFF_);

    hipMemcpyAsync(x, src, NB, hipMemcpyDeviceToDevice, stream);

    for (int blk = 0; blk < 2; ++blk) {
        hipMemcpyAsync(resid, x, NB, hipMemcpyDeviceToDevice, stream);
        const int shift = blk & 1;
        for (int i = 0; i < 2; ++i) {
            const int lid = blk * 2 + i;
            hipMemcpyAsync(ident, x, NB, hipMemcpyDeviceToDevice, stream);

            prep_kernel<<<EG, 256, 0, stream>>>(x, pos + (size_t)lid * NV * CCH, xq, xb, nel);

            // QK projection: (x+pos) @ [Wq;Wk]^T  -> N x 512 bf16
            gemm_bf16_wmma<false, false, true><<<dim3(MB, 8), 256, 0, stream>>>(
                xq, CCH, wqkv + (size_t)lid * 768 * CCH, CCH,
                in_proj_b + lid * 768, nullptr, qkb, 512, NV, CCH);
            // V projection: x @ Wv^T -> N x 256 bf16
            gemm_bf16_wmma<false, false, true><<<dim3(MB, 4), 256, 0, stream>>>(
                xb, CCH, wqkv + (size_t)lid * 768 * CCH + (size_t)512 * CCH, CCH,
                in_proj_b + lid * 768 + 512, nullptr, vb, CCH, NV, CCH);

            attn_kernel<<<NSET_, 256, 0, stream>>>(
                qkb, vb,
                inds  + (size_t)(shift * 2 + i) * NSET_ * SETSZ,
                masks + (size_t)(shift * 2 + i) * NSET_ * SETSZ,
                attnb);

            // out-proj (permutation commutes with row GEMM) -> f32
            gemm_bf16_wmma<false, true, false><<<dim3(MB, 4), 256, 0, stream>>>(
                attnb, CCH, wo + (size_t)lid * CCH * CCH, CCH,
                out_b + lid * CCH, projf, nullptr, CCH, NV, CCH);

            ln_add_kernel<<<LG, 256, 0, stream>>>(
                x, projf, n1_g + lid * CCH, n1_b + lid * CCH, x, xb, NV);

            // FFN1 + exact GELU -> N x 1024 bf16
            gemm_bf16_wmma<true, false, true><<<dim3(MB, 16), 256, 0, stream>>>(
                xb, CCH, w1 + (size_t)lid * DFF_ * CCH, CCH,
                lin1_b + lid * DFF_, nullptr, hb, DFF_, NV, CCH);
            // FFN2 -> f32
            gemm_bf16_wmma<false, true, false><<<dim3(MB, 4), 256, 0, stream>>>(
                hb, DFF_, w2 + (size_t)lid * CCH * DFF_, DFF_,
                lin2_b + lid * CCH, projf, nullptr, CCH, NV, DFF_);

            ln_add_kernel<<<LG, 256, 0, stream>>>(
                x, projf, n2_g + lid * CCH, n2_b + lid * CCH, x, nullptr, NV);
            ln_add_kernel<<<LG, 256, 0, stream>>>(
                x, ident, enc_g + lid * CCH, enc_b + lid * CCH, x, nullptr, NV);
        }
        float* outp = (blk == 1) ? (float*)d_out : x;
        ln_add_kernel<<<LG, 256, 0, stream>>>(
            x, resid, res_g + blk * CCH, res_b + blk * CCH, outp, nullptr, NV);
    }
}
